// SelfAttention2_45200235823179
// MI455X (gfx1250) — matrix operand order repaired and checked by reading, not compiled, not run on hardware
//
#include <hip/hip_runtime.h>

#define B_    4
#define S_    1024
#define D_    1024
#define H_    16
#define HD_   64
#define OUT_  1024

typedef __attribute__((ext_vector_type(16))) __bf16 v16bf;
typedef __attribute__((ext_vector_type(8)))  float  v8f;

union V16U { v16bf v; uint4 u[2]; };
union P8U  { __bf16 h[8]; uint4 u; };
union P2U  { __bf16 h[2]; unsigned int u; };

// ---------------------------------------------------------------------------
// Kernel 1: projection GEMM, C = X[4096x1024] * W[1024x1024], fp32 -> bf16.
// z = 0: Q (row-major), z = 1: K (row-major), z = 2: V (transposed [B][OUT][S])
// 256 threads = 8 waves (2 x 4); block tile 128(M) x 256(N); wave tile 64x64;
// K chunk = 32; software-pipelined staging; LDS tiles pre-swizzled into WMMA
// fragment layout (each lane's 16x32 fragment = 32 contiguous bytes).
// ---------------------------------------------------------------------------
__global__ __launch_bounds__(256) void proj_kernel(
    const float* __restrict__ Xq, const float* __restrict__ Xk, const float* __restrict__ Xv,
    const float* __restrict__ Wq, const float* __restrict__ Wk, const float* __restrict__ Wv,
    __bf16* __restrict__ Qp, __bf16* __restrict__ Kp, __bf16* __restrict__ Vt)
{
  __shared__ alignas(64) __bf16 As[8 * 512];    // 8 M-tiles, frag-linear (8 KB)
  __shared__ alignas(64) __bf16 Bs[16 * 512];   // 16 N-tiles, frag-linear (16 KB)
  __shared__ alignas(64) __bf16 Ts[8 * 256];    // per-wave 16x16 transpose slice (4 KB)

  const int z = blockIdx.z;
  const float* __restrict__ X = (z == 0) ? Xq : (z == 1) ? Xk : Xv;
  const float* __restrict__ W = (z == 0) ? Wq : (z == 1) ? Wk : Wv;

  const int tid  = threadIdx.x;
  const int lane = tid & 31;
  const int wave = tid >> 5;       // 0..7
  const int wm   = wave >> 2;      // 0..1 -> M-tiles wm*4 .. wm*4+3
  const int wn   = wave & 3;       // 0..3 -> N-tiles wn*4 .. wn*4+3
  const int m0   = blockIdx.x * 128;
  const int n0   = blockIdx.y * 256;

  // staging maps
  const int am  = tid >> 1;            // A: row in tile 0..127
  const int akg = (tid & 1) * 16;      // A: k start (0 or 16)
  const int kp  = tid >> 4;            // B: k-pair 0..15 -> rows 2kp, 2kp+1
  const int n16 = (tid & 15) * 16;     // B: col start
  const int bnt = tid & 15;            // B: destination N-tile

  v8f acc[4][4];
#pragma unroll
  for (int i = 0; i < 4; ++i)
#pragma unroll
    for (int j = 0; j < 4; ++j)
      acc[i][j] = (v8f){0.f,0.f,0.f,0.f,0.f,0.f,0.f,0.f};

  float fA[16], fB[32];
  // ---- preload chunk 0 ----
  {
    const float* srcA = X + (size_t)(m0 + am) * D_ + akg;
#pragma unroll
    for (int q4 = 0; q4 < 4; ++q4) {
      float4 t = ((const float4*)srcA)[q4];
      fA[q4*4+0] = t.x; fA[q4*4+1] = t.y; fA[q4*4+2] = t.z; fA[q4*4+3] = t.w;
    }
    const float* srcB = W + (size_t)(2 * kp) * OUT_ + n0 + n16;
#pragma unroll
    for (int r = 0; r < 2; ++r)
#pragma unroll
      for (int q4 = 0; q4 < 4; ++q4) {
        float4 t = ((const float4*)(srcB + (size_t)r * OUT_))[q4];
        fB[r*16 + q4*4+0] = t.x; fB[r*16 + q4*4+1] = t.y;
        fB[r*16 + q4*4+2] = t.z; fB[r*16 + q4*4+3] = t.w;
      }
  }

  for (int kb = 0; kb < D_; kb += 32) {
    __syncthreads();
    // ---- store staged chunk into fragment-layout LDS ----
    {
      const int mtile = am >> 4, mrow = am & 15;
#pragma unroll
      for (int g = 0; g < 2; ++g) {
        const int k0 = akg + 8 * g;
        const int hbit = (k0 >> 4) & 1;
        const int est  = k0 & 8;
        P8U pk;
#pragma unroll
        for (int i = 0; i < 8; ++i) pk.h[i] = (__bf16)fA[8*g + i];
        *(uint4*)&As[mtile * 512 + (mrow + 16 * hbit) * 16 + est] = pk.u;
      }
      const int e0 = (2 * kp) & 15;
      const int hbit = kp >> 3;
#pragma unroll
      for (int i = 0; i < 16; ++i) {
        P2U p; p.h[0] = (__bf16)fB[i]; p.h[1] = (__bf16)fB[16 + i];
        *(unsigned int*)&Bs[bnt * 512 + (i + 16 * hbit) * 16 + e0] = p.u;
      }
    }
    __syncthreads();
    // ---- kick off next chunk's global loads (overlap with WMMAs) ----
    if (kb + 32 < D_) {
      const float* srcA = X + (size_t)(m0 + am) * D_ + kb + 32 + akg;
#pragma unroll
      for (int q4 = 0; q4 < 4; ++q4) {
        float4 t = ((const float4*)srcA)[q4];
        fA[q4*4+0] = t.x; fA[q4*4+1] = t.y; fA[q4*4+2] = t.z; fA[q4*4+3] = t.w;
      }
      const float* srcB = W + (size_t)(kb + 32 + 2 * kp) * OUT_ + n0 + n16;
#pragma unroll
      for (int r = 0; r < 2; ++r)
#pragma unroll
        for (int q4 = 0; q4 < 4; ++q4) {
          float4 t = ((const float4*)(srcB + (size_t)r * OUT_))[q4];
          fB[r*16 + q4*4+0] = t.x; fB[r*16 + q4*4+1] = t.y;
          fB[r*16 + q4*4+2] = t.z; fB[r*16 + q4*4+3] = t.w;
        }
    }
    // ---- fragment loads + 16 WMMAs ----
    v16bf a[4], bfr[4];
#pragma unroll
    for (int i = 0; i < 4; ++i)
      a[i] = *(const v16bf*)&As[(wm * 4 + i) * 512 + lane * 16];
#pragma unroll
    for (int j = 0; j < 4; ++j)
      bfr[j] = *(const v16bf*)&Bs[(wn * 4 + j) * 512 + lane * 16];
#pragma unroll
    for (int i = 0; i < 4; ++i)
#pragma unroll
      for (int j = 0; j < 4; ++j)
        acc[i][j] = __builtin_amdgcn_wmma_f32_16x16x32_bf16(false, a[i], false, bfr[j],
                                                            (short)0, acc[i][j], false, false);
  }

  // ---- epilogue ----
  const int mm = lane & 15, hb = lane >> 4;
  if (z < 2) {
    // transpose each 16x16 accumulator through LDS -> coalesced b128 stores
    __bf16* dst = (z == 0) ? Qp : Kp;
    __bf16* tsl = &Ts[wave * 256];
#pragma unroll
    for (int i = 0; i < 4; ++i)
#pragma unroll
      for (int j = 0; j < 4; ++j) {
        __syncthreads();
#pragma unroll
        for (int r = 0; r < 8; ++r)
          tsl[(r + 8 * hb) * 16 + mm] = (__bf16)acc[i][j][r];
        __syncthreads();
        const int rr = lane >> 1, hf = lane & 1;
        uint4 d = *(const uint4*)&tsl[rr * 16 + hf * 8];
        const int grow = m0 + wm * 64 + i * 16 + rr;
        const int gcol = n0 + wn * 64 + j * 16 + hf * 8;
        *(uint4*)&dst[(size_t)grow * OUT_ + gcol] = d;
      }
  } else {
    // V transposed: Vt[b][n][s]; 8 consecutive rows pack into one b128 store
    const int bidx = m0 >> 10;
#pragma unroll
    for (int i = 0; i < 4; ++i) {
      const int sb = (m0 & (S_ - 1)) + wm * 64 + i * 16 + 8 * hb;
#pragma unroll
      for (int j = 0; j < 4; ++j) {
        const int n = n0 + wn * 64 + j * 16 + mm;
        P8U pk;
#pragma unroll
        for (int r = 0; r < 8; ++r) pk.h[r] = (__bf16)acc[i][j][r];
        *(uint4*)&Vt[((size_t)bidx * OUT_ + n) * S_ + sb] = pk.u;
      }
    }
  }
}

// ---------------------------------------------------------------------------
// Kernel 2: flash attention. One wave per (batch, head, 32-query block):
// two 16-row q tiles share every K/V fragment load.
// ---------------------------------------------------------------------------
__global__ __launch_bounds__(32) void attn_kernel(
    const __bf16* __restrict__ Qp, const __bf16* __restrict__ Kp,
    const __bf16* __restrict__ Vt, const int* __restrict__ Qlen,
    const int* __restrict__ Vlen, float* __restrict__ out)
{
  __shared__ alignas(64) __bf16 Ps[2 * 512];   // P fragments, 2 KB
  __shared__ alignas(64) float  Os[32 * 64];   // output transpose tile, 8 KB

  const int lane = threadIdx.x & 31;
  const int mm = lane & 15, hb = lane >> 4;
  const int qb = blockIdx.x;           // 32-query block 0..31
  const int h  = blockIdx.y;
  const int b  = blockIdx.z;
  const int q0 = qb * 32;
  const int vlen = Vlen[b];
  const int qlen = Qlen[b];
  const v8f czero = (v8f){0.f,0.f,0.f,0.f,0.f,0.f,0.f,0.f};

  // q fragments: 2 q-tiles x 2 HD-chunks
  v16bf qf[2][2];
#pragma unroll
  for (int qt = 0; qt < 2; ++qt) {
    const __bf16* qrow = Qp + (size_t)(b * S_ + q0 + qt * 16 + mm) * OUT_ + h * HD_;
#pragma unroll
    for (int c = 0; c < 2; ++c) {
      V16U t;
      t.u[0] = *(const uint4*)(qrow + c * 32 + hb * 16);
      t.u[1] = *(const uint4*)(qrow + c * 32 + hb * 16 + 8);
      qf[qt][c] = t.v;
    }
  }

  float mstat[2][8], lstat[2][8];
  v8f acc[2][4];
#pragma unroll
  for (int qt = 0; qt < 2; ++qt) {
#pragma unroll
    for (int j = 0; j < 8; ++j) { mstat[qt][j] = -3e38f; lstat[qt][j] = 0.f; }
#pragma unroll
    for (int nt = 0; nt < 4; ++nt) acc[qt][nt] = czero;
  }

  int kend = q0 + 32;                      // causal bound for the block
  const int kcap = (vlen + 31) & ~31;      // key-length bound
  if (kcap < kend) kend = kcap;

  for (int kb = 0; kb < kend; kb += 32) {
    // ---- K fragments: 2 key tiles x 2 HD chunks ----
    v16bf kf[2][2];
#pragma unroll
    for (int t = 0; t < 2; ++t) {
      const __bf16* krow = Kp + (size_t)(b * S_ + kb + t * 16 + mm) * OUT_ + h * HD_;
      V16U k0, k1;
      k0.u[0] = *(const uint4*)(krow + hb * 16);
      k0.u[1] = *(const uint4*)(krow + hb * 16 + 8);
      k1.u[0] = *(const uint4*)(krow + 32 + hb * 16);
      k1.u[1] = *(const uint4*)(krow + 32 + hb * 16 + 8);
      kf[t][0] = k0.v; kf[t][1] = k1.v;
    }
    // ---- prefetch next chunk (global_prefetch_b8) ----
    if (kb + 32 < kend) {
      __builtin_prefetch(Kp + (size_t)(b * S_ + kb + 32 + mm) * OUT_ + h * HD_, 0, 1);
      __builtin_prefetch(Kp + (size_t)(b * S_ + kb + 48 + mm) * OUT_ + h * HD_, 0, 1);
      __builtin_prefetch(Vt + ((size_t)b * OUT_ + h * HD_ + mm) * S_ + kb + 32, 0, 1);
      __builtin_prefetch(Vt + ((size_t)b * OUT_ + h * HD_ + 32 + mm) * S_ + kb + 32, 0, 1);
    }

    __syncthreads();   // previous iteration's Ps reads complete
#pragma unroll
    for (int qt = 0; qt < 2; ++qt) {
      // ---- scores (K = 64 chained as 2 x 32) ----
      v8f s[2];
#pragma unroll
      for (int t = 0; t < 2; ++t) {
        s[t] = __builtin_amdgcn_wmma_f32_16x16x32_bf16(false, qf[qt][0], false, kf[t][0],
                                                       (short)0, czero, false, false);
        s[t] = __builtin_amdgcn_wmma_f32_16x16x32_bf16(false, qf[qt][1], false, kf[t][1],
                                                       (short)0, s[t], false, false);
      }
      // ---- scale + causal + key-length mask ----
#pragma unroll
      for (int t = 0; t < 2; ++t) {
        const int key = kb + t * 16 + mm;
#pragma unroll
        for (int j = 0; j < 8; ++j) {
          const int qrow = q0 + qt * 16 + j + 8 * hb;
          const float v = s[t][j] * 0.125f;
          s[t][j] = ((key > qrow) || (key >= vlen)) ? -1e12f : v;
        }
      }
      // ---- online softmax ----
      float rmax[8];
#pragma unroll
      for (int j = 0; j < 8; ++j) rmax[j] = fmaxf(s[0][j], s[1][j]);
#pragma unroll
      for (int d = 1; d <= 8; d <<= 1)
#pragma unroll
        for (int j = 0; j < 8; ++j)
          rmax[j] = fmaxf(rmax[j], __shfl_xor(rmax[j], d, 32));
      float alpha[8];
#pragma unroll
      for (int j = 0; j < 8; ++j) {
        const float mnew = fmaxf(mstat[qt][j], rmax[j]);
        alpha[j] = __expf(mstat[qt][j] - mnew);
        mstat[qt][j] = mnew;
      }
#pragma unroll
      for (int t = 0; t < 2; ++t)
#pragma unroll
        for (int j = 0; j < 8; ++j)
          s[t][j] = __expf(s[t][j] - mstat[qt][j]);
      float rsum[8];
#pragma unroll
      for (int j = 0; j < 8; ++j) rsum[j] = s[0][j] + s[1][j];
#pragma unroll
      for (int d = 1; d <= 8; d <<= 1)
#pragma unroll
        for (int j = 0; j < 8; ++j)
          rsum[j] += __shfl_xor(rsum[j], d, 32);
#pragma unroll
      for (int j = 0; j < 8; ++j) lstat[qt][j] = lstat[qt][j] * alpha[j] + rsum[j];
#pragma unroll
      for (int nt = 0; nt < 4; ++nt)
#pragma unroll
        for (int j = 0; j < 8; ++j)
          acc[qt][nt][j] *= alpha[j];
      // ---- P: C layout -> A layout via LDS scatter ----
#pragma unroll
      for (int j = 0; j < 8; ++j) {
        const int m = j + 8 * hb;
#pragma unroll
        for (int t = 0; t < 2; ++t) {
          const int k = mm + 16 * t;
          const int owner = m + 16 * ((k >> 4) & 1);
          const int e = k & 15;
          Ps[qt * 512 + owner * 16 + e] = (__bf16)s[t][j];
        }
      }
    }
    __syncthreads();
    const v16bf pf0 = *(const v16bf*)&Ps[lane * 16];
    const v16bf pf1 = *(const v16bf*)&Ps[512 + lane * 16];
    // ---- PV: V fragments contiguous (transposed Vt), shared by both q tiles ----
#pragma unroll
    for (int nt = 0; nt < 4; ++nt) {
      const __bf16* vp = Vt + ((size_t)b * OUT_ + h * HD_ + nt * 16 + mm) * S_ + kb + hb * 16;
      V16U vf;
      vf.u[0] = *(const uint4*)(vp);
      vf.u[1] = *(const uint4*)(vp + 8);
      acc[0][nt] = __builtin_amdgcn_wmma_f32_16x16x32_bf16(false, pf0, false, vf.v,
                                                           (short)0, acc[0][nt], false, false);
      acc[1][nt] = __builtin_amdgcn_wmma_f32_16x16x32_bf16(false, pf1, false, vf.v,
                                                           (short)0, acc[1][nt], false, false);
    }
  }

  // ---- epilogue: normalize, query-length mask, transpose via LDS, coalesced f4 stores
#pragma unroll
  for (int qt = 0; qt < 2; ++qt)
#pragma unroll
    for (int nt = 0; nt < 4; ++nt)
#pragma unroll
      for (int j = 0; j < 8; ++j) {
        const int row = qt * 16 + j + 8 * hb;
        float o = acc[qt][nt][j] / lstat[qt][j];
        if (q0 + row >= qlen) o = 0.f;
        Os[row * 64 + nt * 16 + mm] = o;
      }
  __syncthreads();
#pragma unroll
  for (int q = 0; q < 16; ++q) {
    const int idx = q * 32 + lane;         // float4 index 0..511
    const int row = idx >> 4, c4 = idx & 15;
    float4 d = *(const float4*)&Os[idx * 4];
    *(float4*)&out[(size_t)(b * S_ + q0 + row) * OUT_ + h * HD_ + c4 * 4] = d;
  }
}

// ---------------------------------------------------------------------------
extern "C" void kernel_launch(void* const* d_in, const int* in_sizes, int n_in,
                              void* d_out, int out_size, void* d_ws, size_t ws_size,
                              hipStream_t stream) {
  const float* Qs = (const float*)d_in[0];
  const float* Ks = (const float*)d_in[1];
  const float* Vs = (const float*)d_in[2];
  const float* WQ = (const float*)d_in[3];
  const float* WK = (const float*)d_in[4];
  const float* WV = (const float*)d_in[5];
  const int* Qlen = (const int*)d_in[6];
  const int* Vlen = (const int*)d_in[7];
  float* out = (float*)d_out;

  // workspace: Qp (8MB) | Kp (8MB) | Vt (8MB), all bf16
  char* ws = (char*)d_ws;
  __bf16* Qp = (__bf16*)(ws);
  __bf16* Kp = (__bf16*)(ws + (size_t)(B_ * S_) * OUT_ * 2);
  __bf16* Vt = (__bf16*)(ws + (size_t)2 * (B_ * S_) * OUT_ * 2);

  dim3 g1((B_ * S_) / 128, OUT_ / 256, 3);
  proj_kernel<<<g1, 256, 0, stream>>>(Qs, Ks, Vs, WQ, WK, WV, Qp, Kp, Vt);

  dim3 g2(S_ / 32, H_, B_);
  attn_kernel<<<g2, 32, 0, stream>>>(Qp, Kp, Vt, Qlen, Vlen, out);
}
